// WanTransformerBlock_VSA_61598420959299
// MI455X (gfx1250) — compile-verified, matrix operand order checked by reading, not run on hardware
//
#include <hip/hip_runtime.h>
#include <math.h>

// WanTransformerBlock_VSA on gfx1250: bf16 WMMA GEMMs + TDM/LDS-tiled flash attention.
#define DIM     1536
#define S_LEN   2048
#define L2_LEN  512
#define NHEAD   12
#define HDIM    128
#define FFN_DIM 8960
#define POOL    64
#define NBLK    32     // S_LEN / POOL
#define TKEY    32     // attention key tile

typedef __attribute__((ext_vector_type(8)))  float          v8f;
typedef __attribute__((ext_vector_type(16))) __bf16         v16bf;
typedef __attribute__((ext_vector_type(8)))  unsigned short v8u;
typedef __attribute__((ext_vector_type(4)))  unsigned int   u32x4;
typedef __attribute__((ext_vector_type(8)))  int            i32x8;
typedef __attribute__((ext_vector_type(4)))  int            i32x4;

union AFrag { v16bf bf; v8u u[2]; };

__device__ __forceinline__ unsigned short f2bf(float f) {
  union { float f; unsigned u; } c; c.f = f;
  unsigned u = c.u;
  u += 0x7FFFu + ((u >> 16) & 1u);          // round-to-nearest-even
  return (unsigned short)(u >> 16);
}

#if __has_builtin(__builtin_amdgcn_cvt_pk_bf16_f32)
typedef __attribute__((ext_vector_type(2))) __bf16 v2bf;
__device__ __forceinline__ unsigned pack_bf16(float lo, float hi) {
  union { v2bf v; unsigned u; } c;
  c.v = __builtin_amdgcn_cvt_pk_bf16_f32(lo, hi);   // V_CVT_PK_BF16_F32
  return c.u;
}
#else
__device__ __forceinline__ unsigned pack_bf16(float lo, float hi) {
  return (unsigned)f2bf(lo) | ((unsigned)f2bf(hi) << 16);
}
#endif

__device__ __forceinline__ float fast_tanh(float x) {
#if __has_builtin(__builtin_amdgcn_tanhf)
  return __builtin_amdgcn_tanhf(x);         // gfx1250 V_TANH_F32
#else
  return tanhf(x);
#endif
}

// ---------------------------------------------------------------------------
// Tensor Data Mover: DMA a 2D fp32 tile (tileW x tileH elems, row stride in
// elems) from global memory into LDS. Descriptor per CDNA5 ISA 8.3/8.4.
// This toolchain exposes the 6-operand builtin:
//   (u32x4 group0, i32x8 group1, i32x4 group2, i32x4 group3, i32x8, i32 cpol)
// Issue from one wave; complete with s_wait_tensorcnt + workgroup barrier.
// ---------------------------------------------------------------------------
#if __has_builtin(__builtin_amdgcn_tensor_load_to_lds) && __has_builtin(__builtin_amdgcn_s_wait_tensorcnt)
#define USE_TDM 1
__device__ __forceinline__ void tdm_load_2d_f32(unsigned ldsAddr, const float* gptr,
                                                unsigned tileW, unsigned tileH,
                                                unsigned rowStrideElems) {
  unsigned long long ga = (unsigned long long)(uintptr_t)gptr;
  u32x4 g0;
  g0[0] = 1u;                                               // count=1, user D#
  g0[1] = ldsAddr;                                          // lds_addr (bytes)
  g0[2] = (unsigned)(ga & 0xFFFFFFFFu);                     // global_addr[31:0]
  g0[3] = (unsigned)((ga >> 32) & 0x01FFFFFFu) | (2u << 30);// global_addr[56:32] | type=2
  i32x8 g1;
  g1[0] = (int)(2u << 16);                                  // data_size=2 (4 bytes)
  g1[1] = (int)((tileW & 0xFFFFu) << 16);                   // tensor_dim0[15:0]
  g1[2] = (int)((tileW >> 16) | ((tileH & 0xFFFFu) << 16)); // t_dim0 hi | tensor_dim1 lo
  g1[3] = (int)((tileH >> 16) | (tileW << 16));             // t_dim1 hi | tile_dim0
  g1[4] = (int)(tileH & 0xFFFFu);                           // tile_dim1 | tile_dim2=0
  g1[5] = (int)rowStrideElems;                              // tensor_dim0_stride[31:0]
  g1[6] = 0;                                                // stride0 hi | stride1 lo
  g1[7] = 0;
  i32x4 z4 = {0, 0, 0, 0};
  i32x8 z8 = {0, 0, 0, 0, 0, 0, 0, 0};
  __builtin_amdgcn_tensor_load_to_lds(g0, g1, z4, z4, z8, 0);
}
#endif

// ---------------------------------------------------------------------------
// GEMM: C[M,N] = A[M,K] @ B[K,N] + bias, optional tanh-GELU.
// 256 threads = 8 waves. Block tile 128x128, BK=64 (two WMMA K-steps/stage).
// Wave grid 4(M) x 2(N); each wave owns 32x64 = 2x4 wmma_f32_16x16x32_bf16.
// Requires: M%128==0, N%128==0, K%64==0, bias != nullptr (true at all sites).
// ---------------------------------------------------------------------------
__launch_bounds__(256)
__global__ void gemm_bf16_wmma(const float* __restrict__ A, const float* __restrict__ B,
                               const float* __restrict__ bias, float* __restrict__ C,
                               int M, int N, int K, int act) {
  __shared__ __align__(16) unsigned short As[128 * 72]; // [m][k], stride 72 (144B, 16B-mult)
  __shared__ __align__(16) unsigned short Bs[128 * 72]; // [n][k] (transposed on stage)

  const int t    = threadIdx.x;
  const int lane = t & 31;
  const int wave = t >> 5;
  const int wm   = wave >> 1;               // 0..3 -> M offset wm*32
  const int wn   = wave & 1;                // 0..1 -> N offset wn*64
  const int m0   = blockIdx.y * 128;
  const int n0   = blockIdx.x * 128;
  const int kb   = (lane & 16) ? 8 : 0;     // 16-bit A/B fragment K sub-offset
  const int mr   = lane & 15;

  v8f zacc = {};
  v8f acc[2][4];
  #pragma unroll
  for (int i = 0; i < 2; ++i)
    #pragma unroll
    for (int j = 0; j < 4; ++j) acc[i][j] = zacc;

  for (int k0 = 0; k0 < K; k0 += 64) {
    // Stage A 128x64 fp32 -> bf16 LDS (float4 global loads, 8B LDS stores)
    #pragma unroll
    for (int i = 0; i < 8; ++i) {
      int idx = i * 256 + t;                // 0..2047 float4 slots
      int r = idx >> 4, q = idx & 15;
      const float4 f = *(const float4*)(A + (size_t)(m0 + r) * K + (k0 + q * 4));
      uint2 p;
      p.x = pack_bf16(f.x, f.y);
      p.y = pack_bf16(f.z, f.w);
      *(uint2*)(&As[r * 72 + q * 4]) = p;
    }
    // Stage B 64x128 fp32 -> bf16 LDS transposed to [n][k]
    #pragma unroll
    for (int i = 0; i < 8; ++i) {
      int idx = i * 256 + t;                // 0..2047
      int n = idx & 127, kq = idx >> 7;     // 4 consecutive k per thread
      const float* bp = B + (size_t)(k0 + kq * 4) * N + (n0 + n);
      float b0 = bp[0];
      float b1 = bp[(size_t)N];
      float b2 = bp[2 * (size_t)N];
      float b3 = bp[3 * (size_t)N];
      uint2 p;
      p.x = pack_bf16(b0, b1);
      p.y = pack_bf16(b2, b3);
      *(uint2*)(&Bs[n * 72 + kq * 4]) = p;
    }
    __syncthreads();

    // Two WMMA K-chunks per stage. Fragment = two contiguous 16B LDS reads.
    // Lanes 0-15: K = {0..7, 16..23}; lanes 16-31: K = {8..15, 24..31}.
    #pragma unroll
    for (int kc = 0; kc < 64; kc += 32) {
      AFrag a[2], b[4];
      #pragma unroll
      for (int mi = 0; mi < 2; ++mi) {
        const unsigned short* p = &As[(wm * 32 + mi * 16 + mr) * 72 + kc];
        a[mi].u[0] = *(const v8u*)(p + kb);
        a[mi].u[1] = *(const v8u*)(p + 16 + kb);
      }
      #pragma unroll
      for (int nj = 0; nj < 4; ++nj) {
        const unsigned short* p = &Bs[(wn * 64 + nj * 16 + mr) * 72 + kc];
        b[nj].u[0] = *(const v8u*)(p + kb);
        b[nj].u[1] = *(const v8u*)(p + 16 + kb);
      }
      #pragma unroll
      for (int mi = 0; mi < 2; ++mi)
        #pragma unroll
        for (int nj = 0; nj < 4; ++nj)
          acc[mi][nj] = __builtin_amdgcn_wmma_f32_16x16x32_bf16(
              false, a[mi].bf, false, b[nj].bf, (short)0, acc[mi][nj], false, false);
    }
    __syncthreads();
  }

  // Epilogue. C/D layout: lane%16 -> N, VGPR r -> M (+8 for lanes 16-31).
  #pragma unroll
  for (int mi = 0; mi < 2; ++mi) {
    #pragma unroll
    for (int nj = 0; nj < 4; ++nj) {
      int mBase = m0 + wm * 32 + mi * 16 + ((lane & 16) ? 8 : 0);
      int nCol  = n0 + wn * 64 + nj * 16 + mr;
      float bb = bias[nCol];
      #pragma unroll
      for (int r = 0; r < 8; ++r) {
        float v = acc[mi][nj][r] + bb;
        if (act == 1) { // gelu(tanh approx)
          float v3 = v * v * v;
          v = 0.5f * v * (1.0f + fast_tanh(0.7978845608028654f * (v + 0.044715f * v3)));
        }
        C[(size_t)(mBase + r) * N + nCol] = v;
      }
    }
  }
}

// ---------------------------------------------------------------------------
// Flash attention: one wave per query row, online softmax, D=128 (4 f32/lane).
// 8 waves/block share one head; K/V staged in LDS in 32-key tiles via the
// Tensor Data Mover (fallback: cooperative float4 loads).
// Requires Skv % 32 == 0 (2048 / 512 / 32).
// ---------------------------------------------------------------------------
__launch_bounds__(256)
__global__ void flash_attn(const float* __restrict__ Q, const float* __restrict__ Kp,
                           const float* __restrict__ Vp, float* __restrict__ O, int Skv) {
  __shared__ __align__(16) float Kls[TKEY * HDIM];
  __shared__ __align__(16) float Vls[TKEY * HDIM];

  const int lane = threadIdx.x & 31;
  const int qi   = blockIdx.x * 8 + (threadIdx.x >> 5);
  const int h    = blockIdx.y;
  const float scale = 0.08838834764831845f; // 1/sqrt(128)

  const float4 qv = ((const float4*)(Q + (size_t)qi * DIM + h * HDIM))[lane];
  float m = -1e30f, l = 0.0f;
  float ax = 0.f, ay = 0.f, az = 0.f, aw = 0.f;

  for (int t0 = 0; t0 < Skv; t0 += TKEY) {
#ifdef USE_TDM
    // Wave 0 DMAs both 32x128 fp32 tiles into LDS via TDM, then waits.
    if (threadIdx.x < 32) {
      tdm_load_2d_f32((unsigned)(uintptr_t)(void*)Kls,
                      Kp + (size_t)t0 * DIM + h * HDIM, HDIM, TKEY, DIM);
      tdm_load_2d_f32((unsigned)(uintptr_t)(void*)Vls,
                      Vp + (size_t)t0 * DIM + h * HDIM, HDIM, TKEY, DIM);
      __builtin_amdgcn_s_wait_tensorcnt(0);
    }
    __syncthreads();
#else
    #pragma unroll
    for (int i = 0; i < 4; ++i) {
      int idx = i * 256 + threadIdx.x;      // 0..1023 float4 slots
      int row = idx >> 5, col = idx & 31;
      ((float4*)Kls)[idx] = ((const float4*)(Kp + (size_t)(t0 + row) * DIM + h * HDIM))[col];
      ((float4*)Vls)[idx] = ((const float4*)(Vp + (size_t)(t0 + row) * DIM + h * HDIM))[col];
    }
    __syncthreads();
#endif

    for (int tk = 0; tk < TKEY; ++tk) {
      const float4 kv = ((const float4*)(Kls + tk * HDIM))[lane];
      float p = qv.x * kv.x + qv.y * kv.y + qv.z * kv.z + qv.w * kv.w;
      #pragma unroll
      for (int off = 16; off >= 1; off >>= 1) p += __shfl_xor(p, off, 32);
      float s    = p * scale;
      float mn   = fmaxf(m, s);
      float ex   = __expf(s - mn);
      float corr = __expf(m - mn);
      l = l * corr + ex;
      const float4 vv = ((const float4*)(Vls + tk * HDIM))[lane];
      ax = ax * corr + ex * vv.x;
      ay = ay * corr + ex * vv.y;
      az = az * corr + ex * vv.z;
      aw = aw * corr + ex * vv.w;
      m = mn;
    }
    __syncthreads();
  }
  float inv = 1.0f / l;
  float4 o; o.x = ax * inv; o.y = ay * inv; o.z = az * inv; o.w = aw * inv;
  ((float4*)(O + (size_t)qi * DIM + h * HDIM))[lane] = o;
}

// ---------------------------------------------------------------------------
// LayerNorm over DIM=1536 per row; out = (x-mean)*rstd*a + b.
// onePlus: a := 1 + aVec[c] (adaLN modulation) else a := aVec[c].
// ---------------------------------------------------------------------------
__launch_bounds__(256)
__global__ void ln_row(const float* __restrict__ x, float* __restrict__ out,
                       const float* __restrict__ aVec, const float* __restrict__ bVec,
                       int onePlus) {
  __shared__ float rs[256], rss[256];
  const int row = blockIdx.x;
  const int t   = threadIdx.x;
  float vals[6];
  float s = 0.f, ss = 0.f;
  #pragma unroll
  for (int j = 0; j < 6; ++j) {
    float v = x[(size_t)row * DIM + t + j * 256];
    vals[j] = v; s += v; ss += v * v;
  }
  rs[t] = s; rss[t] = ss; __syncthreads();
  for (int o = 128; o > 0; o >>= 1) {
    if (t < o) { rs[t] += rs[t + o]; rss[t] += rss[t + o]; }
    __syncthreads();
  }
  float mean = rs[0]  * (1.0f / DIM);
  float var  = rss[0] * (1.0f / DIM) - mean * mean;
  float rstd = rsqrtf(var + 1e-6f);
  #pragma unroll
  for (int j = 0; j < 6; ++j) {
    int c = t + j * 256;
    float a = aVec[c]; if (onePlus) a += 1.0f;
    out[(size_t)row * DIM + c] = (vals[j] - mean) * rstd * a + bVec[c];
  }
}

// ---------------------------------------------------------------------------
// In-place RMS-norm (over DIM) * w, optionally followed by RoPE per head.
// Pairs (2p, 2p+1) within each 128-wide head, cos/sin indexed [row][p%64].
// ---------------------------------------------------------------------------
__launch_bounds__(256)
__global__ void rms_rope(float* __restrict__ x, const float* __restrict__ w,
                         const float* __restrict__ cosT, const float* __restrict__ sinT,
                         int useRope) {
  __shared__ float rss[256];
  const int row = blockIdx.x;
  const int t   = threadIdx.x;
  float x1[3], x2[3];
  float ss = 0.f;
  #pragma unroll
  for (int j = 0; j < 3; ++j) {
    int p = t + j * 256;
    float a = x[(size_t)row * DIM + 2 * p];
    float b = x[(size_t)row * DIM + 2 * p + 1];
    x1[j] = a; x2[j] = b; ss += a * a + b * b;
  }
  rss[t] = ss; __syncthreads();
  for (int o = 128; o > 0; o >>= 1) { if (t < o) rss[t] += rss[t + o]; __syncthreads(); }
  float rms = rsqrtf(rss[0] * (1.0f / DIM) + 1e-6f);
  #pragma unroll
  for (int j = 0; j < 3; ++j) {
    int p = t + j * 256;
    float y1 = x1[j] * rms * w[2 * p];
    float y2 = x2[j] * rms * w[2 * p + 1];
    if (useRope) {
      int i = p & 63;
      float c  = cosT[row * 64 + i];
      float sn = sinT[row * 64 + i];
      float o1 = y1 * c - y2 * sn;
      float o2 = y2 * c + y1 * sn;
      y1 = o1; y2 = o2;
    }
    x[(size_t)row * DIM + 2 * p]     = y1;
    x[(size_t)row * DIM + 2 * p + 1] = y2;
  }
}

__global__ void add_vec(const float* __restrict__ a, const float* __restrict__ b,
                        float* __restrict__ out, int n) {
  int i = blockIdx.x * blockDim.x + threadIdx.x;
  if (i < n) out[i] = a[i] + b[i];
}

// Mean over 64 consecutive rows: out[nb, c] = mean_j in[nb*64+j, c]
__global__ void pool64(const float* __restrict__ in, float* __restrict__ out) {
  int i = blockIdx.x * blockDim.x + threadIdx.x;
  if (i >= NBLK * DIM) return;
  int nb = i / DIM, c = i % DIM;
  float s = 0.f;
  #pragma unroll 4
  for (int j = 0; j < POOL; ++j) s += in[(size_t)(nb * POOL + j) * DIM + c];
  out[i] = s * (1.0f / POOL);
}

// out = fine + coarse[s/64] * g   (elementwise over [S, DIM])
__global__ void combine_coarse(const float* __restrict__ fine, const float* __restrict__ coarse,
                               const float* __restrict__ g, float* __restrict__ out) {
  size_t i = (size_t)blockIdx.x * blockDim.x + threadIdx.x;
  if (i >= (size_t)S_LEN * DIM) return;
  size_t srow = i / DIM; int c = (int)(i % DIM);
  out[i] = fine[i] + coarse[(srow / POOL) * DIM + c] * g[i];
}

// out = base + evec[c] * t   (evec == nullptr -> plain add)
__global__ void axpy_row(const float* __restrict__ base, const float* __restrict__ tv,
                         const float* __restrict__ evec, float* __restrict__ out, size_t n) {
  size_t i = (size_t)blockIdx.x * blockDim.x + threadIdx.x;
  if (i >= n) return;
  float g = evec ? evec[i % DIM] : 1.0f;
  out[i] = base[i] + g * tv[i];
}

// ---------------------------------------------------------------------------
extern "C" void kernel_launch(void* const* d_in, const int* in_sizes, int n_in,
                              void* d_out, int out_size, void* d_ws, size_t ws_size,
                              hipStream_t stream) {
  const float* x      = (const float*)d_in[0];
  const float* enc    = (const float*)d_in[1];
  const float* temb   = (const float*)d_in[2];
  const float* rcos   = (const float*)d_in[3];
  const float* rsin   = (const float*)d_in[4];
  const float* sst    = (const float*)d_in[5];
  const float* Wq = (const float*)d_in[6];  const float* bq = (const float*)d_in[7];
  const float* Wk = (const float*)d_in[8];  const float* bk = (const float*)d_in[9];
  const float* Wv = (const float*)d_in[10]; const float* bv = (const float*)d_in[11];
  const float* Wg = (const float*)d_in[12]; const float* bg = (const float*)d_in[13];
  const float* Wo = (const float*)d_in[14]; const float* bo = (const float*)d_in[15];
  const float* nq_w   = (const float*)d_in[16];
  const float* nk_w   = (const float*)d_in[17];
  const float* saln_w = (const float*)d_in[18];
  const float* saln_b = (const float*)d_in[19];
  const float* cWq = (const float*)d_in[20]; const float* cbq = (const float*)d_in[21];
  const float* cWk = (const float*)d_in[22]; const float* cbk = (const float*)d_in[23];
  const float* cWv = (const float*)d_in[24]; const float* cbv = (const float*)d_in[25];
  const float* cWo = (const float*)d_in[26]; const float* cbo = (const float*)d_in[27];
  const float* cnq_w = (const float*)d_in[28];
  const float* cnk_w = (const float*)d_in[29];
  const float* W1 = (const float*)d_in[30]; const float* b1 = (const float*)d_in[31];
  const float* W2 = (const float*)d_in[32]; const float* b2 = (const float*)d_in[33];

  const size_t SD = (size_t)S_LEN * DIM;
  float* ws = (float*)d_ws;
  float* e      = ws; ws += 6 * DIM;
  float* nbuf   = ws; ws += SD;        // n / n2 / n3
  float* qbuf   = ws; ws += SD;        // q, later cq
  float* kbuf   = ws; ws += SD;        // k, later ck
  float* vbuf   = ws; ws += SD;        // v, later cv
  float* gbuf   = ws; ws += SD;        // g
  float* attn   = ws; ws += SD;        // attention outputs
  float* tmp    = ws; ws += SD;        // GEMM outputs (ao / co / ffn2)
  float* hidden = ws; ws += SD;
  float* qc = ws; ws += (size_t)NBLK * DIM;
  float* kc = ws; ws += (size_t)NBLK * DIM;
  float* vc = ws; ws += (size_t)NBLK * DIM;
  float* cc = ws; ws += (size_t)NBLK * DIM;
  float* ffn = ws; // S_LEN * FFN_DIM

  const dim3 blk(256);
  const dim3 gSD((unsigned)((SD + 255) / 256));
  const dim3 gProj(DIM / 128, S_LEN / 128);     // [2048 x 1536] output tiles
  const dim3 gProj512(DIM / 128, L2_LEN / 128); // [512 x 1536]
  const dim3 gFFN1(FFN_DIM / 128, S_LEN / 128); // [2048 x 8960]

  // e = scale_shift_table + temb ; rows: 0 shift, 1 scale, 2 gate, 3 c_shift, 4 c_scale, 5 c_gate
  add_vec<<<dim3((6 * DIM + 255) / 256), blk, 0, stream>>>(sst, temb, e, 6 * DIM);

  // --- Self attention ---
  ln_row<<<dim3(S_LEN), blk, 0, stream>>>(x, nbuf, e + DIM, e, 1);
  gemm_bf16_wmma<<<gProj, blk, 0, stream>>>(nbuf, Wq, bq, qbuf, S_LEN, DIM, DIM, 0);
  gemm_bf16_wmma<<<gProj, blk, 0, stream>>>(nbuf, Wk, bk, kbuf, S_LEN, DIM, DIM, 0);
  gemm_bf16_wmma<<<gProj, blk, 0, stream>>>(nbuf, Wv, bv, vbuf, S_LEN, DIM, DIM, 0);
  gemm_bf16_wmma<<<gProj, blk, 0, stream>>>(nbuf, Wg, bg, gbuf, S_LEN, DIM, DIM, 0);
  rms_rope<<<dim3(S_LEN), blk, 0, stream>>>(qbuf, nq_w, rcos, rsin, 1);
  rms_rope<<<dim3(S_LEN), blk, 0, stream>>>(kbuf, nk_w, rcos, rsin, 1);

  flash_attn<<<dim3(S_LEN / 8, NHEAD), blk, 0, stream>>>(qbuf, kbuf, vbuf, attn, S_LEN);

  pool64<<<dim3((NBLK * DIM + 255) / 256), blk, 0, stream>>>(qbuf, qc);
  pool64<<<dim3((NBLK * DIM + 255) / 256), blk, 0, stream>>>(kbuf, kc);
  pool64<<<dim3((NBLK * DIM + 255) / 256), blk, 0, stream>>>(vbuf, vc);
  flash_attn<<<dim3(NBLK / 8, NHEAD), blk, 0, stream>>>(qc, kc, vc, cc, NBLK);
  combine_coarse<<<gSD, blk, 0, stream>>>(attn, cc, gbuf, attn);

  gemm_bf16_wmma<<<gProj, blk, 0, stream>>>(attn, Wo, bo, tmp, S_LEN, DIM, DIM, 0);
  axpy_row<<<gSD, blk, 0, stream>>>(x, tmp, e + 2 * DIM, hidden, SD);

  // --- Cross attention ---
  ln_row<<<dim3(S_LEN), blk, 0, stream>>>(hidden, nbuf, saln_w, saln_b, 0);
  gemm_bf16_wmma<<<gProj, blk, 0, stream>>>(nbuf, cWq, cbq, qbuf, S_LEN, DIM, DIM, 0);
  rms_rope<<<dim3(S_LEN), blk, 0, stream>>>(qbuf, cnq_w, rcos, rsin, 0);
  gemm_bf16_wmma<<<gProj512, blk, 0, stream>>>(enc, cWk, cbk, kbuf, L2_LEN, DIM, DIM, 0);
  rms_rope<<<dim3(L2_LEN), blk, 0, stream>>>(kbuf, cnk_w, rcos, rsin, 0);
  gemm_bf16_wmma<<<gProj512, blk, 0, stream>>>(enc, cWv, cbv, vbuf, L2_LEN, DIM, DIM, 0);

  flash_attn<<<dim3(S_LEN / 8, NHEAD), blk, 0, stream>>>(qbuf, kbuf, vbuf, attn, L2_LEN);

  gemm_bf16_wmma<<<gProj, blk, 0, stream>>>(attn, cWo, cbo, tmp, S_LEN, DIM, DIM, 0);
  axpy_row<<<gSD, blk, 0, stream>>>(hidden, tmp, nullptr, hidden, SD);

  // --- FFN ---
  ln_row<<<dim3(S_LEN), blk, 0, stream>>>(hidden, nbuf, e + 4 * DIM, e + 3 * DIM, 1);
  gemm_bf16_wmma<<<gFFN1, blk, 0, stream>>>(nbuf, W1, b1, ffn, S_LEN, FFN_DIM, DIM, 1);
  gemm_bf16_wmma<<<gProj, blk, 0, stream>>>(ffn, W2, b2, tmp, S_LEN, DIM, FFN_DIM, 0);
  axpy_row<<<gSD, blk, 0, stream>>>(hidden, tmp, e + 5 * DIM, (float*)d_out, SD);
}
